// FullLobPredModel_3702261809908
// MI455X (gfx1250) — matrix operand order repaired and checked by reading, not compiled, not run on hardware
//
#include <hip/hip_runtime.h>

// ---------------------------------------------------------------------------
// Types for CDNA5 WMMA (wave32): A/B = 16 bf16 halves (8 VGPRs), C/D = 8 f32.
// ---------------------------------------------------------------------------
typedef __attribute__((ext_vector_type(16))) __bf16 bf16x16;
typedef __attribute__((ext_vector_type(8)))  __bf16 bf16x8;
typedef __attribute__((ext_vector_type(8)))  float  f32x8;

#define TM 64
#define TN 64
#define TK 32
#define LDS_K (TK + 8)   // 80B rows: 16B-aligned fragment loads, bank-staggered

// ---------------------------------------------------------------------------
// gfx1250 async global->LDS copy (ASYNCcnt-tracked DMA; CDNA5 ISA).
// lds_off = wave-relative LDS byte address (low 32 bits of generic pointer).
// ---------------------------------------------------------------------------
__device__ __forceinline__ void async_copy_b128(unsigned lds_off, const void* gaddr) {
  asm volatile("global_load_async_to_lds_b128 %0, %1, off"
               :: "v"(lds_off), "v"(gaddr)
               : "memory");
}
__device__ __forceinline__ void wait_async0() {
  asm volatile("s_wait_asynccnt 0" ::: "memory");
}

// ---------------------------------------------------------------------------
// Bulk fp32 -> bf16 conversion (one-time per operand; keeps conversion out of
// the GEMM hot loop and halves GEMM global traffic). n8 = elements / 8.
// ---------------------------------------------------------------------------
__global__ __launch_bounds__(256) void cvt_f32_bf16(
    const float* __restrict__ src, __bf16* __restrict__ dst, long n8)
{
  long i = (long)blockIdx.x * 256 + threadIdx.x;
  if (i >= n8) return;
  const float4* s = (const float4*)src + 2 * i;
  float4 f0 = s[0], f1 = s[1];
  bf16x8 r;
  r[0] = (__bf16)f0.x; r[1] = (__bf16)f0.y; r[2] = (__bf16)f0.z; r[3] = (__bf16)f0.w;
  r[4] = (__bf16)f1.x; r[5] = (__bf16)f1.y; r[6] = (__bf16)f1.z; r[7] = (__bf16)f1.w;
  *(bf16x8*)(dst + 8 * i) = r;
}

// ---------------------------------------------------------------------------
// GEMM: C(M,N) [+]= A(M,K) * B(K,N); A,B bf16, C fp32.
//   ATRANS=false: A element (m,k) = A[m*lda + k]; A tile staged via
//                 global_load_async_to_lds_b128 (pure 16B copies).
//   ATRANS=true : A element (m,k) = A[k*lda + m]  (m contiguous; projections)
//   B element (k,n) = B[k*ldb + n]; staged transposed through VGPRs.
//   bias_mode: 0 none, 1 per-column bias[n], 2 per-row bias[m]
//   accum:     0 store, 1 add, 2 subtract (C -= A*B)
// Block = 256 threads = 8 wave32, block tile 64x64, K-step 32.
// Wave w: rows wm=(w>>1)*16, cols wn=(w&1)*16 and wn+32 -> one A fragment
// feeds two WMMAs. Requires M%64==0, N%64==0, K%32==0, lda/ldb%8==0.
// ---------------------------------------------------------------------------
template <bool ATRANS>
__global__ __launch_bounds__(256) void gemm_bf16_wmma(
    const __bf16* __restrict__ A, int lda,
    const __bf16* __restrict__ B, int ldb,
    float* __restrict__ C, int ldc,
    const float* __restrict__ bias, int bias_mode,
    int M, int N, int K, int accum)
{
  __shared__ __bf16 As[TM][LDS_K];   // (m, k)
  __shared__ __bf16 Bt[TN][LDS_K];   // (n, k)  (B staged transposed)
  const int tid  = threadIdx.x;
  const int wave = tid >> 5;
  const int lane = tid & 31;
  const int m0 = blockIdx.x * TM;
  const int n0 = blockIdx.y * TN;
  const int wm = (wave >> 1) << 4;   // 0/16/32/48
  const int wn = (wave & 1) << 4;    // 0/16
  const int hi  = lane >> 4;
  const int l15 = lane & 15;

  f32x8 acc0 = {0.f,0.f,0.f,0.f,0.f,0.f,0.f,0.f};
  f32x8 acc1 = {0.f,0.f,0.f,0.f,0.f,0.f,0.f,0.f};

  // Per-thread staging coordinates.
  const int am  = tid >> 2;            // A (NN): row,  8 bf16 along k
  const int ak0 = (tid & 3) << 3;
  const int ak  = tid >> 3;            // A (TN): k row, 8 bf16 along m
  const int am0 = (tid & 7) << 3;
  const int bk  = tid >> 3;            // B: k row, 8 bf16 along n
  const int bn0 = (tid & 7) << 3;
  const unsigned ldsA =
      (unsigned)(unsigned long long)(const void*)&As[am][ak0];

  for (int k0 = 0; k0 < K; k0 += TK) {
    if (!ATRANS) {
      // A tile 64x32: async DMA copy straight into LDS (no VGPR round-trip).
      async_copy_b128(ldsA, A + (long)(m0 + am) * lda + (k0 + ak0));
    } else {
      // Transposed A: contiguous along m; scatter into As rows.
      bf16x8 v = *(const bf16x8*)(A + (long)(k0 + ak) * lda + (m0 + am0));
#pragma unroll
      for (int j = 0; j < 8; ++j) As[am0 + j][ak] = v[j];
    }
    {
      // B tile 32x64: contiguous along n; scatter into Bt (n,k) rows.
      bf16x8 v = *(const bf16x8*)(B + (long)(k0 + bk) * ldb + (n0 + bn0));
#pragma unroll
      for (int j = 0; j < 8; ++j) Bt[bn0 + j][bk] = v[j];
    }
    // Prefetch next K tile (global_prefetch_b8).
    if (k0 + TK < K) {
      if (!ATRANS)
        __builtin_prefetch(A + (long)(m0 + am) * lda + (k0 + TK + ak0), 0, 1);
      else
        __builtin_prefetch(A + (long)(k0 + TK + ak) * lda + (m0 + am0), 0, 1);
      __builtin_prefetch(B + (long)(k0 + TK + bk) * ldb + (n0 + bn0), 0, 1);
    }
    if (!ATRANS) wait_async0();        // drain ASYNCcnt before the barrier
    __syncthreads();

    // Fragments per ISA 7.12.2 16-bit layouts.
    const __bf16* arow  = &As[wm + l15][0];
    const __bf16* brow0 = &Bt[wn + l15][0];
    const __bf16* brow1 = &Bt[wn + 32 + l15][0];
    bf16x8 alo = *(const bf16x8*)(arow  + (hi << 3));
    bf16x8 ahi = *(const bf16x8*)(arow  + 16 + (hi << 3));
    bf16x8 b0l = *(const bf16x8*)(brow0 + (hi << 3));
    bf16x8 b0h = *(const bf16x8*)(brow0 + 16 + (hi << 3));
    bf16x8 b1l = *(const bf16x8*)(brow1 + (hi << 3));
    bf16x8 b1h = *(const bf16x8*)(brow1 + 16 + (hi << 3));
    bf16x16 a, b0, b1;
#pragma unroll
    for (int j = 0; j < 8; ++j) {
      a[j]  = alo[j]; a[8 + j]  = ahi[j];
      b0[j] = b0l[j]; b0[8 + j] = b0h[j];
      b1[j] = b1l[j]; b1[8 + j] = b1h[j];
    }
    acc0 = __builtin_amdgcn_wmma_f32_16x16x32_bf16(false, a, false, b0,
                                                   (short)0, acc0, false, false);
    acc1 = __builtin_amdgcn_wmma_f32_16x16x32_bf16(false, a, false, b1,
                                                   (short)0, acc1, false, false);
    __syncthreads();
  }

  // C/D layout: VGPR v, lanes 0-15 -> M=v; lanes 16-31 -> M=8+v.
  const int col0 = n0 + wn + l15;
  const int col1 = col0 + 32;
#pragma unroll
  for (int v = 0; v < 8; ++v) {
    int row = m0 + wm + (hi << 3) + v;
    float r0 = acc0[v], r1 = acc1[v];
    if (bias_mode == 1)      { r0 += bias[col0]; r1 += bias[col1]; }
    else if (bias_mode == 2) { float br = bias[row]; r0 += br; r1 += br; }
    float* c0 = C + (long)row * ldc + col0;
    float* c1 = C + (long)row * ldc + col1;
    if (accum == 0)      { *c0 = r0;       *c1 = r1;       }
    else if (accum == 1) { *c0 = *c0 + r0; *c1 = *c1 + r1; }
    else                 { *c0 = *c0 - r0; *c1 = *c1 - r1; }
  }
}

// ---------------------------------------------------------------------------
// ZOH discretization: Lam_bar = exp(Lam*dt), coef = (Lam_bar - 1)/Lam.
// ---------------------------------------------------------------------------
__global__ void ssm_prep_lambda(const float* __restrict__ Lam_re,
                                const float* __restrict__ Lam_im,
                                const float* __restrict__ log_step,
                                float2* __restrict__ lam_bar,
                                float2* __restrict__ coef, int P)
{
  int p = blockIdx.x * blockDim.x + threadIdx.x;
  if (p >= P) return;
  float lr = Lam_re[p], li = Lam_im[p];
  float dt = __expf(log_step[p]);
  float er = __expf(lr * dt);
  float cb = __cosf(li * dt), sb = __sinf(li * dt);
  float2 lb = make_float2(er * cb, er * sb);
  lam_bar[p] = lb;
  float nr = lb.x - 1.0f, ni = lb.y;
  float den = lr * lr + li * li;
  coef[p] = make_float2((nr * lr + ni * li) / den, (ni * lr - nr * li) / den);
}

// B_bar^T (H,P) and C^T (P,H), emitted directly in bf16 GEMM layouts.
__global__ void ssm_prep_mats(const float* __restrict__ B_re,   // (P,H)
                              const float* __restrict__ B_im,
                              const float* __restrict__ C_re,   // (H,P)
                              const float* __restrict__ C_im,
                              const float2* __restrict__ coef,
                              __bf16* __restrict__ Bt_re,       // (H,P)
                              __bf16* __restrict__ Bt_im,
                              __bf16* __restrict__ Ct_re,       // (P,H)
                              __bf16* __restrict__ Ct_im,
                              int P, int H)
{
  int idx = blockIdx.x * blockDim.x + threadIdx.x;
  if (idx >= P * H) return;
  int p = idx / H, h = idx % H;
  float2 cf = coef[p];
  float br = B_re[(long)p * H + h], bi = B_im[(long)p * H + h];
  Bt_re[(long)h * P + p] = (__bf16)(cf.x * br - cf.y * bi);
  Bt_im[(long)h * P + p] = (__bf16)(cf.x * bi + cf.y * br);
  Ct_re[(long)p * H + h] = (__bf16)C_re[(long)h * P + p];
  Ct_im[(long)p * H + h] = (__bf16)C_im[(long)h * P + p];
}

// ---------------------------------------------------------------------------
// 3-phase segmented scan of x_t = A*x_{t-1} + Bu_t (A constant per channel).
// Threads map to channels (t == p, P == 256): all Bu/xs accesses are fully
// coalesced rows. S = 256 segments, Lseg = L/256. xs written in bf16 (its
// only consumer is the following GEMM).
// ---------------------------------------------------------------------------
__global__ __launch_bounds__(256) void scan_seg_reduce(
    const float* __restrict__ Bu_re, const float* __restrict__ Bu_im, // (L,P)
    const float2* __restrict__ lam_bar,
    float2* __restrict__ carry,                                       // (S,P)
    int Lseg, int P)
{
  const int p = threadIdx.x;
  const int s = blockIdx.x;
  const float2 A = lam_bar[p];
  const long base = (long)s * Lseg;
  float xr = 0.f, xi = 0.f;
  for (int i = 0; i < Lseg; ++i) {
    long off = (base + i) * P + p;
    float br = Bu_re[off], bi = Bu_im[off];
    float nr = A.x * xr - A.y * xi + br;
    float ni = A.x * xi + A.y * xr + bi;
    xr = nr; xi = ni;
  }
  carry[(long)s * P + p] = make_float2(xr, xi);
}

__global__ __launch_bounds__(256) void scan_carry(
    float2* __restrict__ carry, const float2* __restrict__ lam_bar,
    int S, int Lseg, int P)
{
  const int p = threadIdx.x;
  const float2 A = lam_bar[p];
  float ar = 1.f, ai = 0.f;                    // A^Lseg
  for (int i = 0; i < Lseg; ++i) {
    float nr = ar * A.x - ai * A.y;
    float ni = ar * A.y + ai * A.x;
    ar = nr; ai = ni;
  }
  float cr = 0.f, ci = 0.f;                    // exclusive carry
  for (int s = 0; s < S; ++s) {
    long off = (long)s * P + p;
    float2 cs = carry[off];
    carry[off] = make_float2(cr, ci);
    float nr = ar * cr - ai * ci + cs.x;
    float ni = ar * ci + ai * cr + cs.y;
    cr = nr; ci = ni;
  }
}

__global__ __launch_bounds__(256) void scan_seg_apply(
    const float* __restrict__ Bu_re, const float* __restrict__ Bu_im,
    const float2* __restrict__ lam_bar,
    const float2* __restrict__ carry,
    __bf16* __restrict__ xs_re, __bf16* __restrict__ xs_im,
    int Lseg, int P)
{
  const int p = threadIdx.x;
  const int s = blockIdx.x;
  const float2 A = lam_bar[p];
  const float2 c0 = carry[(long)s * P + p];
  const long base = (long)s * Lseg;
  float xr = c0.x, xi = c0.y;
  for (int i = 0; i < Lseg; ++i) {
    long off = (base + i) * P + p;
    float br = Bu_re[off], bi = Bu_im[off];
    float nr = A.x * xr - A.y * xi + br;
    float ni = A.x * xi + A.y * xr + bi;
    xr = nr; xi = ni;
    xs_re[off] = (__bf16)xr;
    xs_im[off] = (__bf16)xi;
  }
}

// ---------------------------------------------------------------------------
// Fused: y = Y + u*D; g = gelu_tanh(y); v = u + g; Out = LayerNorm(v)*g + b.
// One block (256 threads) per row; H <= 512.
// ---------------------------------------------------------------------------
__global__ __launch_bounds__(256) void post_block(
    const float* __restrict__ X, const float* __restrict__ Y,
    const float* __restrict__ Dp, const float* __restrict__ g,
    const float* __restrict__ b, float* __restrict__ Out, int H)
{
  const int row = blockIdx.x;
  const int t = threadIdx.x;
  __shared__ float red[8];
  __shared__ float stats[2];

  float v[2] = {0.f, 0.f};
  float sum = 0.f;
#pragma unroll
  for (int i = 0; i < 2; ++i) {
    int h = t + (i << 8);
    if (h < H) {
      float u = X[(long)row * H + h];
      float yv = Y[(long)row * H + h] + u * Dp[h];
      float c = 0.7978845608028654f * (yv + 0.044715f * yv * yv * yv);
      float ge = 0.5f * yv * (1.0f + tanhf(c));
      v[i] = u + ge;
      sum += v[i];
    }
  }
  for (int off = 16; off > 0; off >>= 1) sum += __shfl_down(sum, off, 32);
  if ((t & 31) == 0) red[t >> 5] = sum;
  __syncthreads();
  if (t == 0) { float s = 0.f; for (int i = 0; i < 8; ++i) s += red[i]; stats[0] = s; }
  __syncthreads();
  float mu = stats[0] / (float)H;

  float d2 = 0.f;
#pragma unroll
  for (int i = 0; i < 2; ++i) {
    int h = t + (i << 8);
    if (h < H) { float dd = v[i] - mu; d2 += dd * dd; }
  }
  for (int off = 16; off > 0; off >>= 1) d2 += __shfl_down(d2, off, 32);
  if ((t & 31) == 0) red[t >> 5] = d2;
  __syncthreads();
  if (t == 0) { float s = 0.f; for (int i = 0; i < 8; ++i) s += red[i]; stats[1] = s; }
  __syncthreads();
  float inv = rsqrtf(stats[1] / (float)H + 1e-6f);

#pragma unroll
  for (int i = 0; i < 2; ++i) {
    int h = t + (i << 8);
    if (h < H) Out[(long)row * H + h] = (v[i] - mu) * inv * g[h] + b[h];
  }
}

// ---------------------------------------------------------------------------
// Mean-pool over rows + decoder + log_softmax. Single block.
// ---------------------------------------------------------------------------
__global__ __launch_bounds__(256) void head_kernel(
    const float* __restrict__ X,      // (R, Hm)
    const float* __restrict__ W,      // (Hm, 3)
    const float* __restrict__ bb,     // (3)
    float* __restrict__ out, int R, int Hm)
{
  __shared__ float mean[512];
  __shared__ float logits[3];
  const int t = threadIdx.x;
  for (int j = t; j < Hm; j += 256) {
    float s = 0.f;
    for (int i = 0; i < R; ++i) s += X[(long)i * Hm + j];
    mean[j] = s / (float)R;
  }
  __syncthreads();
  if (t < 3) {
    float s = bb[t];
    for (int j = 0; j < Hm; ++j) s += mean[j] * W[j * 3 + t];
    logits[t] = s;
  }
  __syncthreads();
  if (t == 0) {
    float m = fmaxf(logits[0], fmaxf(logits[1], logits[2]));
    float e0 = __expf(logits[0] - m);
    float e1 = __expf(logits[1] - m);
    float e2 = __expf(logits[2] - m);
    float lse = logf(e0 + e1 + e2) + m;
    out[0] = logits[0] - lse;
    out[1] = logits[1] - lse;
    out[2] = logits[2] - lse;
  }
}

// ---------------------------------------------------------------------------
// Host-side orchestration
// ---------------------------------------------------------------------------
struct LayerPtrs {
  const float *ln_b, *ln_g, *B_im, *B_re, *C_im, *C_re, *D, *Lam_im, *Lam_re, *log_step;
};
static inline LayerPtrs layer_at(void* const* d_in, int i) {
  LayerPtrs lp;
  lp.ln_b    = (const float*)d_in[i + 0];
  lp.ln_g    = (const float*)d_in[i + 1];
  lp.B_im    = (const float*)d_in[i + 2];
  lp.B_re    = (const float*)d_in[i + 3];
  lp.C_im    = (const float*)d_in[i + 4];
  lp.C_re    = (const float*)d_in[i + 5];
  lp.D       = (const float*)d_in[i + 6];
  lp.Lam_im  = (const float*)d_in[i + 7];
  lp.Lam_re  = (const float*)d_in[i + 8];
  lp.log_step= (const float*)d_in[i + 9];
  return lp;
}

static inline void launch_cvt(const float* src, __bf16* dst, long n, hipStream_t s) {
  long n8 = n / 8;
  cvt_f32_bf16<<<(int)((n8 + 255) / 256), 256, 0, s>>>(src, dst, n8);
}

struct Ws {
  float *X, *Y, *Bu_re, *Bu_im, *CC;
  __bf16 *xs_re, *xs_im, *Xbf, *Wbf;
  __bf16 *Bt_re, *Bt_im, *Ct_re, *Ct_im;
  float2 *lam_bar, *coef, *carry;
};

static void run_layer(const float* Xin, float* Xout, float* Ywork,
                      const LayerPtrs& lp, int L, int H, const Ws& w, hipStream_t s)
{
  const int P = 256;
  const int S = 256;
  const int Lseg = L / S;
  launch_cvt(Xin, w.Xbf, (long)L * H, s);
  ssm_prep_lambda<<<1, 256, 0, s>>>(lp.Lam_re, lp.Lam_im, lp.log_step, w.lam_bar, w.coef, P);
  ssm_prep_mats<<<(P * H + 255) / 256, 256, 0, s>>>(lp.B_re, lp.B_im, lp.C_re, lp.C_im,
                                                    w.coef, w.Bt_re, w.Bt_im, w.Ct_re, w.Ct_im,
                                                    P, H);
  dim3 g1(L / TM, P / TN);
  gemm_bf16_wmma<false><<<g1, 256, 0, s>>>(w.Xbf, H, w.Bt_re, P, w.Bu_re, P, nullptr, 0, L, P, H, 0);
  gemm_bf16_wmma<false><<<g1, 256, 0, s>>>(w.Xbf, H, w.Bt_im, P, w.Bu_im, P, nullptr, 0, L, P, H, 0);
  scan_seg_reduce<<<S, 256, 0, s>>>(w.Bu_re, w.Bu_im, w.lam_bar, w.carry, Lseg, P);
  scan_carry<<<1, 256, 0, s>>>(w.carry, w.lam_bar, S, Lseg, P);
  scan_seg_apply<<<S, 256, 0, s>>>(w.Bu_re, w.Bu_im, w.lam_bar, w.carry, w.xs_re, w.xs_im, Lseg, P);
  dim3 g2(L / TM, H / TN);
  gemm_bf16_wmma<false><<<g2, 256, 0, s>>>(w.xs_re, P, w.Ct_re, H, Ywork, H, nullptr, 0, L, H, P, 0);
  gemm_bf16_wmma<false><<<g2, 256, 0, s>>>(w.xs_im, P, w.Ct_im, H, Ywork, H, nullptr, 0, L, H, P, 2);
  post_block<<<L, 256, 0, s>>>(Xin, Ywork, lp.D, lp.ln_g, lp.ln_b, Xout, H);
}

extern "C" void kernel_launch(void* const* d_in, const int* in_sizes, int n_in,
                              void* d_out, int out_size, void* d_ws, size_t ws_size,
                              hipStream_t stream)
{
  (void)in_sizes; (void)n_in; (void)out_size; (void)ws_size;
  const int Lm = 16384, Lb = 16384, dm = 512, P = 256;

  const float* x_m       = (const float*)d_in[0];
  const float* x_b       = (const float*)d_in[1];
  const float* msg_enc_W = (const float*)d_in[4];
  const float* msg_enc_b = (const float*)d_in[5];
  LayerPtrs msg_l0  = layer_at(d_in, 6);
  LayerPtrs msg_l1  = layer_at(d_in, 16);
  const float* mid_W = (const float*)d_in[26];
  const float* mid_b = (const float*)d_in[27];
  LayerPtrs book_post = layer_at(d_in, 28);
  LayerPtrs book_pre  = layer_at(d_in, 38);
  const float* f_enc_W = (const float*)d_in[48];
  const float* f_enc_b = (const float*)d_in[49];
  LayerPtrs f_l0 = layer_at(d_in, 50);
  LayerPtrs f_l1 = layer_at(d_in, 60);
  const float* msg_out_W  = (const float*)d_in[70];
  const float* msg_out_b  = (const float*)d_in[71];
  const float* book_out_W = (const float*)d_in[72];
  const float* book_out_b = (const float*)d_in[73];
  const float* dec_W = (const float*)d_in[74];
  const float* dec_b = (const float*)d_in[75];

  const long LH = (long)Lm * dm;   // 8,388,608
  const long LP = (long)Lm * P;    // 4,194,304

  float* ws = (float*)d_ws;
  Ws w;
  w.X     = ws;
  w.Y     = ws + LH;
  w.Bu_re = ws + 2 * LH;
  w.Bu_im = w.Bu_re + LP;
  w.xs_re = (__bf16*)(w.Bu_im + LP);
  w.xs_im = w.xs_re + LP;
  w.Xbf   = w.xs_im + LP;
  w.Wbf   = w.Xbf + LH;
  float* pbase = (float*)(w.Wbf + LH);
  w.lam_bar = (float2*)pbase;                  // 256 float2
  w.coef    = w.lam_bar + P;                   // 256 float2
  w.Bt_re = (__bf16*)(pbase + 1024);
  w.Bt_im = w.Bt_re + (long)dm * P;
  w.Ct_re = w.Bt_im + (long)dm * P;
  w.Ct_im = w.Ct_re + (long)dm * P;
  w.carry = (float2*)(w.Ct_im + (long)dm * P); // 256*256 float2
  w.CC    = (float*)(w.carry + 256 * 256);     // (512, 1024) concat buffer

  // ---------------- message path ----------------
  launch_cvt(x_m, w.Xbf, (long)Lm * 64, stream);
  launch_cvt(msg_enc_W, w.Wbf, 64L * dm, stream);
  {
    dim3 g(Lm / TM, dm / TN);
    gemm_bf16_wmma<false><<<g, 256, 0, stream>>>(w.Xbf, 64, w.Wbf, dm,
                                                 w.X, dm, msg_enc_b, 1, Lm, dm, 64, 0);
  }
  run_layer(w.X, w.X, w.Y, msg_l0, Lm, dm, w, stream);
  run_layer(w.X, w.X, w.Y, msg_l1, Lm, dm, w, stream);
  // xm_f[i,j] = sum_l W[l,i]*xm[l,j] + b[i]  -> CC[:, 0:512]
  launch_cvt(w.X, w.Xbf, LH, stream);
  launch_cvt(msg_out_W, w.Wbf, LH, stream);
  {
    dim3 g(dm / TM, dm / TN);
    gemm_bf16_wmma<true><<<g, 256, 0, stream>>>(w.Wbf, dm, w.Xbf, dm,
                                                w.CC, 2 * dm, msg_out_b, 2, dm, dm, Lm, 0);
  }

  // ---------------- book path ----------------
  run_layer(x_b, w.X, w.Y, book_pre, Lb, 128, w, stream);
  launch_cvt(w.X, w.Xbf, (long)Lb * 128, stream);
  launch_cvt(mid_W, w.Wbf, 128L * dm, stream);
  {
    dim3 g(Lb / TM, dm / TN);
    gemm_bf16_wmma<false><<<g, 256, 0, stream>>>(w.Xbf, 128, w.Wbf, dm,
                                                 w.Y, dm, mid_b, 1, Lb, dm, 128, 0);
  }
  run_layer(w.Y, w.Y, w.X, book_post, Lb, dm, w, stream);
  // xb_f -> CC[:, 512:1024]
  launch_cvt(w.Y, w.Xbf, LH, stream);
  launch_cvt(book_out_W, w.Wbf, LH, stream);
  {
    dim3 g(dm / TM, dm / TN);
    gemm_bf16_wmma<true><<<g, 256, 0, stream>>>(w.Wbf, dm, w.Xbf, dm,
                                                w.CC + dm, 2 * dm, book_out_b, 2, dm, dm, Lb, 0);
  }

  // ---------------- fused path ----------------
  launch_cvt(w.CC, w.Xbf, (long)dm * 2 * dm, stream);
  launch_cvt(f_enc_W, w.Wbf, (long)2 * dm * dm, stream);
  {
    dim3 g(dm / TM, dm / TN);
    gemm_bf16_wmma<false><<<g, 256, 0, stream>>>(w.Xbf, 2 * dm, w.Wbf, dm,
                                                 w.X, dm, f_enc_b, 1, dm, dm, 2 * dm, 0);
  }
  run_layer(w.X, w.X, w.Y, f_l0, dm, dm, w, stream);
  run_layer(w.X, w.X, w.Y, f_l1, dm, dm, w, stream);

  head_kernel<<<1, 256, 0, stream>>>(w.X, dec_W, dec_b, (float*)d_out, dm, dm);
}
